// PointNetCurvature_33603824123947
// MI455X (gfx1250) — compile-verified, matrix operand order checked by reading, not run on hardware
//
#include <hip/hip_runtime.h>
#include <hip/hip_bf16.h>

// ---------------------------------------------------------------------------
// Types for CDNA5 WMMA (wave32): bf16 A/B fragments (16 elems/lane = 8 VGPRs),
// f32 C/D accumulator (8 elems/lane = 8 VGPRs).
// ---------------------------------------------------------------------------
typedef __attribute__((ext_vector_type(16))) __bf16      v16bf;
typedef __attribute__((ext_vector_type(8)))  float       v8f;
typedef __attribute__((ext_vector_type(4)))  unsigned    u32x4;

union FragBF16 { u32x4 q[2]; v16bf v; };

__device__ __forceinline__ unsigned short f2bf(float f) {
    unsigned u = __float_as_uint(f);
    unsigned r = u + 0x7FFFu + ((u >> 16) & 1u);   // round-to-nearest-even
    return (unsigned short)(r >> 16);
}

// Order-preserving float <-> uint encoding (for atomicMax/Min on floats).
__device__ __forceinline__ unsigned fenc(float f) {
    unsigned u = __float_as_uint(f);
    return (u & 0x80000000u) ? ~u : (u ^ 0x80000000u);
}
__device__ __forceinline__ float fdec(unsigned u) {
    unsigned v = (u & 0x80000000u) ? (u ^ 0x80000000u) : ~u;
    return __uint_as_float(v);
}

#define BATCH 32
#define NPTS  2048
#define MROWS (BATCH * NPTS)   // 65536

// ---------------------------------------------------------------------------
// Init: zero the stat accumulators, set pooled max/min identities.
// ---------------------------------------------------------------------------
__global__ void k_init(float* statsF, int nf, unsigned* pmax, unsigned* pmin, int np) {
    int i = blockIdx.x * blockDim.x + threadIdx.x;
    if (i < nf) statsF[i] = 0.f;
    if (i < np) { pmax[i] = 0u; pmin[i] = 0xFFFFFFFFu; }
}

// ---------------------------------------------------------------------------
// Curvature: one workgroup per 256 points of one batch sample; all 2048
// points of the sample cached in LDS (24 KB). Each thread brute-forces 5-NN
// with a 6-deep sorted register list (self distance 0 occupies slot 0).
// ---------------------------------------------------------------------------
__global__ __launch_bounds__(256)
void k_curvature(const float* __restrict__ x, float* __restrict__ h0) {
    __shared__ float sp[NPTS * 3];
    const int bb = blockIdx.y;
    const float* xb = x + (size_t)bb * NPTS * 3;
    for (int i = threadIdx.x; i < NPTS * 3; i += 256) sp[i] = xb[i];
    __syncthreads();

    const int p = blockIdx.x * 256 + threadIdx.x;
    const float px = sp[p * 3 + 0], py = sp[p * 3 + 1], pz = sp[p * 3 + 2];

    float b0 = 3.402823466e38f, b1 = b0, b2 = b0, b3 = b0, b4 = b0, b5 = b0;
    for (int j = 0; j < NPTS; ++j) {
        float dx = px - sp[j * 3 + 0];
        float dy = py - sp[j * 3 + 1];
        float dz = pz - sp[j * 3 + 2];
        float d = sqrtf(fmaxf(dx * dx + dy * dy + dz * dz, 0.f));
        if (d < b5) {
            b5 = d; float t;
            if (b5 < b4) { t = b4; b4 = b5; b5 = t;
              if (b4 < b3) { t = b3; b3 = b4; b4 = t;
                if (b3 < b2) { t = b2; b2 = b3; b3 = t;
                  if (b2 < b1) { t = b1; b1 = b2; b2 = t;
                    if (b1 < b0) { t = b0; b0 = b1; b1 = t; } } } } }
        }
    }
    float dmean = (b1 + b2 + b3 + b4 + b5) * 0.2f;  // drop self (b0 == 0)
    float curv  = 1.0f / (1e-8f + dmean);
    float4 o; o.x = px; o.y = py; o.z = pz; o.w = curv;
    *(float4*)(h0 + ((size_t)bb * NPTS + p) * 4) = o;
}

// ---------------------------------------------------------------------------
// Conv1 (4 -> 64): K=4, pure VALU (WMMA useless at K=4 for bf16 shapes).
// ---------------------------------------------------------------------------
__global__ void k_conv1(const float4* __restrict__ h0, const float* __restrict__ w,
                        const float* __restrict__ bias, float* __restrict__ out, int M) {
    int i = blockIdx.x * blockDim.x + threadIdx.x;
    if (i >= M * 64) return;
    int m = i >> 6, n = i & 63;
    float4 pv = h0[m];
    const float* wr = w + n * 4;
    out[i] = pv.x * wr[0] + pv.y * wr[1] + pv.z * wr[2] + pv.w * wr[3] + bias[n];
}

// ---------------------------------------------------------------------------
// Per-channel sum / sumsq over [M, C] (C power of two), LDS-first reduction.
// ---------------------------------------------------------------------------
__global__ __launch_bounds__(256)
void k_colstats(const float* __restrict__ act, int total, int Cmask,
                float* __restrict__ sum, float* __restrict__ sq) {
    __shared__ float sS[128], sQ[128];
    int C = Cmask + 1;
    if ((int)threadIdx.x < C) { sS[threadIdx.x] = 0.f; sQ[threadIdx.x] = 0.f; }
    __syncthreads();
    for (int i = blockIdx.x * blockDim.x + threadIdx.x; i < total;
         i += gridDim.x * blockDim.x) {
        float v = act[i];
        int c = i & Cmask;
        atomicAdd(&sS[c], v);
        atomicAdd(&sQ[c], v * v);
    }
    __syncthreads();
    if ((int)threadIdx.x < C) {
        atomicAdd(&sum[threadIdx.x], sS[threadIdx.x]);
        atomicAdd(&sq[threadIdx.x],  sQ[threadIdx.x]);
    }
}

__global__ void k_bnfinal(const float* sum, const float* sq, const float* g,
                          const float* b, float* scale, float* shift, int C, float invN) {
    int c = blockIdx.x * blockDim.x + threadIdx.x;
    if (c >= C) return;
    float m   = sum[c] * invN;
    float var = sq[c] * invN - m * m;
    float inv = rsqrtf(var + 1e-5f);
    float sc  = g[c] * inv;
    scale[c] = sc;
    shift[c] = b[c] - m * sc;
}

__global__ void k_normrelu(const float* __restrict__ act, const float* __restrict__ scale,
                           const float* __restrict__ shift, unsigned short* __restrict__ out,
                           int total, int Cmask) {
    int i = blockIdx.x * blockDim.x + threadIdx.x;
    if (i >= total) return;
    int c = i & Cmask;
    out[i] = f2bf(fmaxf(act[i] * scale[c] + shift[c], 0.f));
}

__global__ void k_f32_to_bf16(const float* __restrict__ src,
                              unsigned short* __restrict__ dst, int n) {
    int i = blockIdx.x * blockDim.x + threadIdx.x;
    if (i < n) dst[i] = f2bf(src[i]);
}

// ---------------------------------------------------------------------------
// bf16 WMMA GEMM: C[M,N] = A[M,K] * W[N,K]^T (+bias).
// Block = 8 waves, tile 128x128; W panel staged once in LDS via the CDNA5
// async global->LDS path (GLOBAL_LOAD_ASYNC_TO_LDS_B128, ASYNCcnt), bypassing
// VGPRs. Fragment layouts follow CDNA5 ISA (lanes 0-15: K 0-7/16-23,
// lanes 16-31: K 8-15/24-31; contiguous 16B chunks per lane).
// FUSED path (conv3): no C store; reduces per-channel sum/sumsq and
// per-(batch,channel) max/min straight from the accumulators.
// ---------------------------------------------------------------------------
template <int KDIM, bool FUSED>
__global__ __launch_bounds__(256)
void k_gemm_bf16(const unsigned short* __restrict__ A,
                 const unsigned short* __restrict__ W,
                 const float* __restrict__ bias,
                 float* __restrict__ Cout, int M, int N,
                 float* __restrict__ sumc, float* __restrict__ sqc,
                 unsigned* __restrict__ pmax, unsigned* __restrict__ pmin,
                 int pointsPerBatch) {
    __shared__ unsigned short sW[128 * KDIM];
    __shared__ float    rSum[128], rSq[128];
    __shared__ unsigned rMax[128], rMin[128];

    const int tid = threadIdx.x;
    const int nt0 = blockIdx.y * 128;

    { // stage weight panel (rows nt0..nt0+127 of [N,KDIM] are contiguous)
      // via async global->LDS copies: no VGPR round trip, tracked by ASYNCcnt.
        const int chunks = 128 * KDIM / 8;          // 16-byte chunks
        const char* srcBase = (const char*)(W + (size_t)nt0 * KDIM);
        // flat LDS aperture keeps the wave-relative LDS byte offset in addr[31:0]
        const unsigned ldsBase = (unsigned)(uintptr_t)(&sW[0]);
        for (int i = tid; i < chunks; i += 256) {
            unsigned    ldsAddr = ldsBase + (unsigned)i * 16u;
            const char* gAddr   = srcBase + (size_t)i * 16;
            asm volatile("global_load_async_to_lds_b128 %0, %1, off"
                         :: "v"(ldsAddr), "v"(gAddr) : "memory");
        }
        asm volatile("s_wait_asynccnt 0x0" ::: "memory");
    }
    if (FUSED && tid < 128) {
        rSum[tid] = 0.f; rSq[tid] = 0.f; rMax[tid] = 0u; rMin[tid] = 0xFFFFFFFFu;
    }
    __syncthreads();

    const int wid   = tid >> 5;
    const int lane  = tid & 31;
    const int lhalf = lane >> 4;       // 0 | 1
    const int l16   = lane & 15;
    const int mrow  = blockIdx.x * 128 + wid * 16 + l16;
    const unsigned short* Arow = A + (size_t)mrow * KDIM;

    v8f acc[8];
    #pragma unroll
    for (int t = 0; t < 8; ++t)
        #pragma unroll
        for (int r = 0; r < 8; ++r) acc[t][r] = 0.f;

    #pragma unroll
    for (int k = 0; k < KDIM; k += 32) {
        const int off = k + lhalf * 8;
        FragBF16 a;
        a.q[0] = *(const u32x4*)(Arow + off);
        a.q[1] = *(const u32x4*)(Arow + off + 16);
        #pragma unroll
        for (int t = 0; t < 8; ++t) {
            const unsigned short* Bp = sW + (size_t)(t * 16 + l16) * KDIM + off;
            FragBF16 bfr;
            bfr.q[0] = *(const u32x4*)(Bp);
            bfr.q[1] = *(const u32x4*)(Bp + 16);
            acc[t] = __builtin_amdgcn_wmma_f32_16x16x32_bf16(
                false, a.v, false, bfr.v, (short)0, acc[t], false, false);
        }
    }

    if (!FUSED) {
        const int grow0 = blockIdx.x * 128 + wid * 16 + lhalf * 8;
        #pragma unroll
        for (int t = 0; t < 8; ++t) {
            const int gn = nt0 + t * 16 + l16;
            const float bn = bias[gn];
            #pragma unroll
            for (int r = 0; r < 8; ++r)
                Cout[(size_t)(grow0 + r) * N + gn] = acc[t][r] + bn;
        }
    } else {
        #pragma unroll
        for (int t = 0; t < 8; ++t) {
            const int nl = t * 16 + l16;
            const float bn = bias[nt0 + nl];
            float s = 0.f, ss = 0.f;
            float mx = -3.402823466e38f, mn = 3.402823466e38f;
            #pragma unroll
            for (int r = 0; r < 8; ++r) {
                float v = acc[t][r] + bn;
                s += v; ss += v * v;
                mx = fmaxf(mx, v); mn = fminf(mn, v);
            }
            s  += __shfl_xor(s, 16);
            ss += __shfl_xor(ss, 16);
            mx = fmaxf(mx, __shfl_xor(mx, 16));
            mn = fminf(mn, __shfl_xor(mn, 16));
            if (lane < 16) {
                atomicAdd(&rSum[nl], s);
                atomicAdd(&rSq[nl], ss);
                atomicMax(&rMax[nl], fenc(mx));
                atomicMin(&rMin[nl], fenc(mn));
            }
        }
        __syncthreads();
        if (tid < 128) {
            const int gn = nt0 + tid;
            const int bb = (blockIdx.x * 128) / pointsPerBatch;  // tile within 1 batch
            atomicAdd(&sumc[gn], rSum[tid]);
            atomicAdd(&sqc[gn],  rSq[tid]);
            atomicMax(&pmax[(size_t)bb * N + gn], rMax[tid]);
            atomicMin(&pmin[(size_t)bb * N + gn], rMin[tid]);
        }
    }
}

// ---------------------------------------------------------------------------
// pooled[b,c] = relu(BN3(maxpool(conv3))) using monotone-affine identity:
// max relu(sc*x+sh) = relu(sc * (sc>=0 ? max x : min x) + sh).
// ---------------------------------------------------------------------------
__global__ void k_pool_relu(const float* __restrict__ sum, const float* __restrict__ sq,
                            const float* __restrict__ g, const float* __restrict__ b,
                            const unsigned* __restrict__ pmax, const unsigned* __restrict__ pmin,
                            float* __restrict__ pooled, int C, float invN) {
    int i = blockIdx.x * blockDim.x + threadIdx.x;
    if (i >= BATCH * C) return;
    int c = i & (C - 1);
    float m   = sum[c] * invN;
    float var = sq[c] * invN - m * m;
    float inv = rsqrtf(var + 1e-5f);
    float sc  = g[c] * inv;
    float sh  = b[c] - m * sc;
    float base = (sc >= 0.f) ? fdec(pmax[i]) : fdec(pmin[i]);
    pooled[i] = fmaxf(sc * base + sh, 0.f);
}

// ---------------------------------------------------------------------------
// Tiny FC layers (<= 17 MFLOP): plain VALU.
// ---------------------------------------------------------------------------
__global__ void k_fc(const float* __restrict__ in, const float* __restrict__ w,
                     const float* __restrict__ bias, float* __restrict__ out,
                     int R, int K, int N) {
    int i = blockIdx.x * blockDim.x + threadIdx.x;
    if (i >= R * N) return;
    int r = i / N, n = i - r * N;
    const float* ir = in + (size_t)r * K;
    const float* wr = w + (size_t)n * K;
    float s = bias[n];
    for (int k = 0; k < K; ++k) s += ir[k] * wr[k];
    out[i] = s;
}

__global__ void k_bnrow_relu(float* __restrict__ act, const float* __restrict__ g,
                             const float* __restrict__ b, int R, int C) {
    int c = blockIdx.x * blockDim.x + threadIdx.x;
    if (c >= C) return;
    float s = 0.f, ss = 0.f;
    for (int r = 0; r < R; ++r) { float v = act[r * C + c]; s += v; ss += v * v; }
    float m   = s / R;
    float var = ss / R - m * m;
    float inv = rsqrtf(var + 1e-5f);
    float sc  = g[c] * inv, sh = b[c] - m * sc;
    for (int r = 0; r < R; ++r)
        act[r * C + c] = fmaxf(act[r * C + c] * sc + sh, 0.f);
}

__global__ __launch_bounds__(64)
void k_out_lsm(const float* __restrict__ h, const float* __restrict__ w,
               const float* __restrict__ bias, float* __restrict__ y) {
    __shared__ float lg[40];
    int r = blockIdx.x, n = threadIdx.x;
    if (n < 40) {
        const float* hr = h + r * 256;
        const float* wr = w + n * 256;
        float s = bias[n];
        for (int k = 0; k < 256; ++k) s += hr[k] * wr[k];
        lg[n] = s;
    }
    __syncthreads();
    if (n < 40) {
        float m = -3.402823466e38f;
        for (int j = 0; j < 40; ++j) m = fmaxf(m, lg[j]);
        float s = 0.f;
        for (int j = 0; j < 40; ++j) s += expf(lg[j] - m);
        y[r * 40 + n] = lg[n] - m - logf(s);
    }
}

// ---------------------------------------------------------------------------
extern "C" void kernel_launch(void* const* d_in, const int* in_sizes, int n_in,
                              void* d_out, int out_size, void* d_ws, size_t ws_size,
                              hipStream_t stream) {
    const float* x    = (const float*)d_in[0];
    const float* c1w  = (const float*)d_in[1];  const float* c1b  = (const float*)d_in[2];
    const float* bn1g = (const float*)d_in[3];  const float* bn1b = (const float*)d_in[4];
    const float* c2w  = (const float*)d_in[5];  const float* c2b  = (const float*)d_in[6];
    const float* bn2g = (const float*)d_in[7];  const float* bn2b = (const float*)d_in[8];
    const float* c3w  = (const float*)d_in[9];  const float* c3b  = (const float*)d_in[10];
    const float* bn3g = (const float*)d_in[11]; const float* bn3b = (const float*)d_in[12];
    const float* f1w  = (const float*)d_in[13]; const float* f1b  = (const float*)d_in[14];
    const float* bf1g = (const float*)d_in[15]; const float* bf1b = (const float*)d_in[16];
    const float* f2w  = (const float*)d_in[17]; const float* f2b  = (const float*)d_in[18];
    const float* bf2g = (const float*)d_in[19]; const float* bf2b = (const float*)d_in[20];
    const float* ow   = (const float*)d_in[21]; const float* ob   = (const float*)d_in[22];
    float* out = (float*)d_out;

    const int M = MROWS;

    // -- workspace carve-up (256B aligned) --
    char* wp = (char*)d_ws;
    auto alloc = [&](size_t bytes) -> void* {
        void* r = (void*)wp;
        wp += (bytes + 255) & ~(size_t)255;
        return r;
    };
    float*          h0    = (float*)         alloc((size_t)M * 4 * 4);
    float*          a1    = (float*)         alloc((size_t)M * 64 * 4);
    unsigned short* a1n   = (unsigned short*)alloc((size_t)M * 64 * 2);
    float*          a2    = (float*)         alloc((size_t)M * 128 * 4);
    unsigned short* a2n   = (unsigned short*)alloc((size_t)M * 128 * 2);
    unsigned short* w2bf  = (unsigned short*)alloc((size_t)128 * 64 * 2);
    unsigned short* w3bf  = (unsigned short*)alloc((size_t)1024 * 128 * 2);
    float*          stats = (float*)         alloc((size_t)(64*4 + 128*4 + 1024*2) * 4);
    unsigned*       pmax  = (unsigned*)      alloc((size_t)BATCH * 1024 * 4);
    unsigned*       pmin  = (unsigned*)      alloc((size_t)BATCH * 1024 * 4);
    float*          pooled= (float*)         alloc((size_t)BATCH * 1024 * 4);
    float*          z1    = (float*)         alloc((size_t)BATCH * 512 * 4);
    float*          z2    = (float*)         alloc((size_t)BATCH * 256 * 4);
    (void)ws_size; (void)n_in; (void)in_sizes; (void)out_size;

    float* sum1 = stats;        float* sq1 = sum1 + 64;
    float* sc1  = sq1 + 64;     float* sh1 = sc1 + 64;
    float* sum2 = sh1 + 64;     float* sq2 = sum2 + 128;
    float* sc2  = sq2 + 128;    float* sh2 = sc2 + 128;
    float* sum3 = sh2 + 128;    float* sq3 = sum3 + 1024;
    const int nStats = 64*4 + 128*4 + 1024*2;   // 2816 floats

    const float invM = 1.0f / (float)M;

    // 0: init accumulators (fresh every launch; deterministic)
    k_init<<<(BATCH*1024 + 255)/256, 256, 0, stream>>>(stats, nStats, pmax, pmin, BATCH*1024);

    // 1: curvature -> h0 [M,4]
    k_curvature<<<dim3(NPTS/256, BATCH), 256, 0, stream>>>(x, h0);

    // weights -> bf16
    k_f32_to_bf16<<<(128*64 + 255)/256, 256, 0, stream>>>(c2w, w2bf, 128*64);
    k_f32_to_bf16<<<(1024*128 + 255)/256, 256, 0, stream>>>(c3w, w3bf, 1024*128);

    // 2: conv1 + bias -> a1 [M,64]; BN stats; normalize+relu -> bf16
    k_conv1<<<(M*64)/256, 256, 0, stream>>>((const float4*)h0, c1w, c1b, a1, M);
    k_colstats<<<1024, 256, 0, stream>>>(a1, M*64, 63, sum1, sq1);
    k_bnfinal<<<1, 64, 0, stream>>>(sum1, sq1, bn1g, bn1b, sc1, sh1, 64, invM);
    k_normrelu<<<(M*64)/256, 256, 0, stream>>>(a1, sc1, sh1, a1n, M*64, 63);

    // 3: conv2 via WMMA bf16 -> a2 [M,128]; BN; -> bf16
    k_gemm_bf16<64, false><<<dim3(M/128, 1), 256, 0, stream>>>(
        a1n, w2bf, c2b, a2, M, 128, nullptr, nullptr, nullptr, nullptr, 0);
    k_colstats<<<1024, 256, 0, stream>>>(a2, M*128, 127, sum2, sq2);
    k_bnfinal<<<1, 128, 0, stream>>>(sum2, sq2, bn2g, bn2b, sc2, sh2, 128, invM);
    k_normrelu<<<(M*128)/256, 256, 0, stream>>>(a2, sc2, sh2, a2n, M*128, 127);

    // 4: conv3 via fused WMMA: BN stats + per-batch max/min, no 256MB spill
    k_gemm_bf16<128, true><<<dim3(M/128, 1024/128), 256, 0, stream>>>(
        a2n, w3bf, c3b, nullptr, M, 1024, sum3, sq3, pmax, pmin, NPTS);

    // 5: maxpool + BN3 + relu -> pooled [32,1024]
    k_pool_relu<<<(BATCH*1024)/256, 256, 0, stream>>>(
        sum3, sq3, bn3g, bn3b, pmax, pmin, pooled, 1024, invM);

    // 6: tail MLP (tiny)
    k_fc<<<(BATCH*512)/256, 256, 0, stream>>>(pooled, f1w, f1b, z1, BATCH, 1024, 512);
    k_bnrow_relu<<<2, 256, 0, stream>>>(z1, bf1g, bf1b, BATCH, 512);
    k_fc<<<(BATCH*256)/256, 256, 0, stream>>>(z1, f2w, f2b, z2, BATCH, 512, 256);
    k_bnrow_relu<<<1, 256, 0, stream>>>(z2, bf2g, bf2b, BATCH, 256);
    k_out_lsm<<<BATCH, 64, 0, stream>>>(z2, ow, ob, out);
}
